// KGConstrainedAttention_20203526160555
// MI455X (gfx1250) — matrix operand order repaired and checked by reading, not compiled, not run on hardware
//
#include <hip/hip_runtime.h>

// ---------------------------------------------------------------------------
// Types for CDNA5 WMMA (wave32): v_wmma_f32_16x16x32_bf16
// ---------------------------------------------------------------------------
typedef __attribute__((ext_vector_type(16))) __bf16       v16bf;
typedef __attribute__((ext_vector_type(8)))  float        v8f;
typedef __attribute__((ext_vector_type(4)))  unsigned int v4u;
typedef __attribute__((ext_vector_type(8)))  unsigned int v8u;

union FragU { v16bf v; v4u u[2]; };

__device__ __forceinline__ v16bf ld_frag2(const __bf16* lo, const __bf16* hi) {
  FragU f;
  f.u[0] = *(const v4u*)lo;   // 16B
  f.u[1] = *(const v4u*)hi;   // 16B
  return f.v;
}

// CDNA5 async global->LDS copy (32B per lane), tracked by ASYNCcnt.
__device__ __forceinline__ void async_copy_32B(const __bf16* gsrc, __bf16* ldst) {
  unsigned           loff = (unsigned)(uintptr_t)ldst;        // LDS byte offset
  unsigned long long ga   = (unsigned long long)(uintptr_t)gsrc;
  asm volatile("global_load_async_to_lds_b128 %0, %1, off\n\t"
               "global_load_async_to_lds_b128 %0, %1, off offset:16"
               :: "v"(loff), "v"(ga) : "memory");
}
__device__ __forceinline__ void wait_asynccnt0() {
  asm volatile("s_wait_asynccnt 0" ::: "memory");
}

// ---------------------------------------------------------------------------
// CDNA5 Tensor Data Mover: 2D tile load (global -> LDS), one DMA per wave.
// D# layout per CDNA5 ISA.
// tensor_w/tensor_h in elements; stride in elements; 2-byte elements.
// LDS padding: pad_amount 4 DWORDs (16B) after every 16 DWORDs (64B) so a
// 32-elem (64B) tile row lands on an 80B (40-elem) LDS row stride.
// ---------------------------------------------------------------------------
__device__ __forceinline__ void tdm_load_2d_bf16(
    unsigned lds_off, const void* gptr,
    unsigned tensor_w, unsigned tensor_h,
    unsigned tile_w, unsigned tile_h, unsigned stride_elems)
{
  unsigned long long ga = (unsigned long long)(uintptr_t)gptr;
  v4u g0;
  g0.x = 0x1u;                                      // count=1, user mode
  g0.y = lds_off;                                   // lds_addr  (bits 63:32)
  g0.z = (unsigned)(ga & 0xFFFFFFFFu);              // global_addr[31:0]
  g0.w = (unsigned)((ga >> 32) & 0x01FFFFFFu)       // global_addr[56:32]
       | (2u << 30);                                // type = 2 ("image")
  v8u g1;
  g1.s0 = (1u << 16)        // data_size = 1 -> 2 bytes
        | (1u << 20)        // pad_enable
        | (3u << 22)        // pad_interval code 3 -> every 16 DWORDs (64B)
        | (3u << 25);       // pad_amount  code 3 -> 4 DWORDs (16B)
  g1.s1 = (tensor_w & 0xFFFFu) << 16;               // tensor_dim0[15:0]
  g1.s2 = (tensor_w >> 16) | ((tensor_h & 0xFFFFu) << 16);
  g1.s3 = (tensor_h >> 16) | ((tile_w & 0xFFFFu) << 16);
  g1.s4 = (tile_h & 0xFFFFu);                       // tile_dim1 (tile_dim2=0)
  g1.s5 = stride_elems;                             // tensor_dim0_stride[31:0]
  g1.s6 = 0;
  g1.s7 = 0;
  asm volatile("tensor_load_to_lds %0, %1" :: "s"(g0), "s"(g1) : "memory");
}

// ---------------------------------------------------------------------------
// Problem constants
// ---------------------------------------------------------------------------
#define EMBED  1024
#define NHEADS 16
#define HD     64
#define BATCH  2
#define SEQ    2048
#define MTOT   (BATCH * SEQ)     // 4096 rows for all projections
#define MASKW  (SEQ / 32)        // packed mask words per row (64)

// ---------------------------------------------------------------------------
// Elementwise converts / mask bit-pack
// ---------------------------------------------------------------------------
__global__ void cvt_bf16_kernel(const float* __restrict__ s, __bf16* __restrict__ d, int n) {
  int i = blockIdx.x * blockDim.x + threadIdx.x;
  if (i < n) d[i] = (__bf16)s[i];
}

// Pack kg_mask into bit-words: word[i] bit j = (mask[i*32+j] != 0)
__global__ void mask_pack_kernel(const float* __restrict__ s, unsigned* __restrict__ w, int n) {
  int i = blockIdx.x * blockDim.x + threadIdx.x;
  if (i < n) {
    unsigned long long bal = __ballot(s[i] != 0.0f);   // wave32: low 32 bits
    if ((threadIdx.x & 31) == 0) w[i >> 5] = (unsigned)bal;
  }
}

// ---------------------------------------------------------------------------
// Tiled bf16 GEMM: Y(4096x1024) = (A(4096x1024) @ W(1024x1024) + bias)*oscale
// Block: 256 thr = 8 waves; block tile 128x128; wave tile 32x64 (2x4 WMMA).
// A tile staged by the Tensor Data Mover (TENSORcnt path, one DMA / K-step).
// MODE 0: scatter bf16 into (bh, s, hd)          (Q and K)
// MODE 2: scatter bf16 into (bh, hd, s) (V^T)    (V)
// MODE 3: plain f32 row-major to d_out           (output projection)
// ---------------------------------------------------------------------------
template <int MODE>
__global__ __launch_bounds__(256) void gemm_bf16_kernel(
    const __bf16* __restrict__ A, const __bf16* __restrict__ W,
    const float* __restrict__ bias, float oscale, void* __restrict__ outp)
{
  constexpr int K = EMBED, N = EMBED;
  __shared__ __bf16 alds[128][40];  // A tile, row-major [m][k], pad 8 (TDM pad)
  __shared__ __bf16 wlds[128][40];  // W tile transposed  [n][k], pad 8

  const int tid  = threadIdx.x;
  const int wave = tid >> 5, lane = tid & 31;
  const int lm = lane & 15, h = lane >> 4, h8 = h * 8;
  const int m0 = blockIdx.x * 128, n0 = blockIdx.y * 128;
  const int mt = (wave >> 1) * 32, nt = (wave & 1) * 64;

  v8f z = {};
  v8f acc[2][4];
#pragma unroll
  for (int i = 0; i < 2; ++i)
#pragma unroll
    for (int j = 0; j < 4; ++j) acc[i][j] = z;

  for (int k0 = 0; k0 < K; k0 += 32) {
    // ---- stage A tile via TDM: 128 rows x 32 k, wave 0 issues the DMA ----
    if (wave == 0) {
      tdm_load_2d_bf16((unsigned)(uintptr_t)&alds[0][0],
                       A + (size_t)m0 * K + k0,
                       /*tensor_w=*/K, /*tensor_h=*/MTOT,
                       /*tile_w=*/32, /*tile_h=*/128, /*stride=*/K);
    }
    // ---- stage W tile transposed: [n][k] so B-fragments are contiguous ----
    {
      int kk = tid >> 3;            // 0..31
      int nc = (tid & 7) * 16;      // 0..112
      const __bf16* src = W + (size_t)(k0 + kk) * N + n0 + nc;
      __bf16 tmp[16];
      *(v4u*)&tmp[0] = *(const v4u*)src;
      *(v4u*)&tmp[8] = *(const v4u*)(src + 8);
#pragma unroll
      for (int i = 0; i < 16; ++i) wlds[nc + i][kk] = tmp[i];
    }
    if (k0 + 32 < K) {  // hint next K tile into cache (global_prefetch_b8)
      __builtin_prefetch(W + (size_t)(k0 + 32 + (tid >> 3)) * N + n0, 0, 1);
    }
    if (wave == 0) __builtin_amdgcn_s_wait_tensorcnt(0);  // DMA landed in LDS
    __syncthreads();

    // ---- fragments: A 16x32 (lane half h: K = h*8+j, 16+h*8+j) ----
    v16bf af[2], bf[4];
#pragma unroll
    for (int mi = 0; mi < 2; ++mi) {
      const __bf16* p = &alds[mt + mi * 16 + lm][0];
      af[mi] = ld_frag2(p + h8, p + 16 + h8);
    }
    // ---- B 32x16 (lane n=lm, K = h*8+j, 16+h*8+j) ----
#pragma unroll
    for (int ni = 0; ni < 4; ++ni) {
      const __bf16* p = &wlds[nt + ni * 16 + lm][0];
      bf[ni] = ld_frag2(p + h8, p + 16 + h8);
    }
#pragma unroll
    for (int mi = 0; mi < 2; ++mi)
#pragma unroll
      for (int ni = 0; ni < 4; ++ni)
        acc[mi][ni] = __builtin_amdgcn_wmma_f32_16x16x32_bf16(
            false, af[mi], false, bf[ni], (short)0, acc[mi][ni], false, false);
    __syncthreads();
  }

  // ---- epilogue: C/D layout: VGPR r -> M = r + 8*h, N = lm ----
#pragma unroll
  for (int mi = 0; mi < 2; ++mi)
#pragma unroll
    for (int ni = 0; ni < 4; ++ni) {
      int nn = n0 + nt + ni * 16 + lm;
      float bv = bias[nn];
#pragma unroll
      for (int r = 0; r < 8; ++r) {
        int mm = m0 + mt + mi * 16 + r + h8;
        float val = (acc[mi][ni][r] + bv) * oscale;
        if constexpr (MODE == 3) {
          ((float*)outp)[(size_t)mm * N + nn] = val;
        } else {
          int b = mm >> 11, s = mm & (SEQ - 1);
          int head = nn >> 6, d = nn & (HD - 1);
          __bf16* o = (__bf16*)outp;
          if constexpr (MODE == 0)
            o[((size_t)(b * NHEADS + head) * SEQ + s) * HD + d] = (__bf16)val;
          else  // MODE 2: V transposed (bh, hd, s)
            o[((size_t)(b * NHEADS + head) * HD + d) * SEQ + s] = (__bf16)val;
        }
      }
    }
}

// ---------------------------------------------------------------------------
// Flash-style masked attention with bit-packed KG mask.
// Grid: (SEQ/64, BATCH*NHEADS). Block: 128 thr = 4 waves, wave = 16 queries.
// Q is pre-scaled by 1/sqrt(hd) in the projection. V tile staged once per
// block via GLOBAL_LOAD_ASYNC_TO_LDS (shared by all 4 waves).
// ---------------------------------------------------------------------------
__global__ __launch_bounds__(128) void attn_kernel(
    const __bf16* __restrict__ Q, const __bf16* __restrict__ Kc,
    const __bf16* __restrict__ Vt, const unsigned* __restrict__ Mw,
    __bf16* __restrict__ Out)
{
  __shared__ __bf16 pbuf[4][16][40];  // per-wave P transpose buffer (q x k)
  __shared__ __bf16 vlds[64][40];     // shared V tile: 64 d-rows x 32 keys

  const int tid = threadIdx.x, wave = tid >> 5, lane = tid & 31;
  const int lm = lane & 15, h = lane >> 4, h8 = h * 8;
  const int bh = blockIdx.y, b = bh >> 4, head = bh & 15;
  const int qbase = blockIdx.x * 64 + wave * 16;

  const __bf16*   Qh = Q  + (size_t)bh * SEQ * HD;
  const __bf16*   Kh = Kc + (size_t)bh * SEQ * HD;
  const __bf16*   Vh = Vt + (size_t)bh * HD * SEQ;
  const unsigned* Mh = Mw + (size_t)b * SEQ * MASKW;

  // Q A-fragments over the two 32-wide hd chunks (resident for whole loop)
  v16bf qa[2];
#pragma unroll
  for (int kc = 0; kc < 2; ++kc) {
    const __bf16* p = Qh + (size_t)(qbase + lm) * HD + kc * 32;
    qa[kc] = ld_frag2(p + h8, p + 16 + h8);
  }

  v8f z = {};
  v8f acc[4] = {z, z, z, z};
  float mrun[8], lrun[8];
#pragma unroll
  for (int r = 0; r < 8; ++r) { mrun[r] = -3.0e38f; lrun[r] = 0.0f; }

  for (int kb2 = 0; kb2 < SEQ; kb2 += 32) {
    // ---- async-stage V tile (64 x 32) once per block: 32B per thread ----
    {
      int row = tid >> 1, half = tid & 1;
      const __bf16* src = Vh + (size_t)row * SEQ + kb2 + half * 16;
      async_copy_32B(src, &vlds[row][half * 16]);
    }

    // ---- scores: two 16x16 tiles (keys kb2..+15, kb2+16..+31) ----
    v8f st[2];
#pragma unroll
    for (int ntile = 0; ntile < 2; ++ntile) {
      const __bf16* kp = Kh + (size_t)(kb2 + ntile * 16 + lm) * HD;
      v16bf k0 = ld_frag2(kp + h8,      kp + 16 + h8);   // d = h*8+[0..7], 16+h*8+[0..7]
      v16bf k1 = ld_frag2(kp + 32 + h8, kp + 48 + h8);   // d = 32 + same
      st[ntile] = __builtin_amdgcn_wmma_f32_16x16x32_bf16(
          false, qa[0], false, k0, (short)0, z, false, false);
      st[ntile] = __builtin_amdgcn_wmma_f32_16x16x32_bf16(
          false, qa[1], false, k1, (short)0, st[ntile], false, false);
    }

    // ---- mask via packed bits (one b32 word covers 32 keys; Q pre-scaled) --
    const int wi = kb2 >> 5;
    float sv[2][8];
#pragma unroll
    for (int r = 0; r < 8; ++r) {
      unsigned mbits = Mh[(size_t)(qbase + r + h8) * MASKW + wi];
      sv[0][r] = ((mbits >> lm)        & 1u) ? st[0][r] : -1.0e30f;
      sv[1][r] = ((mbits >> (16 + lm)) & 1u) ? st[1][r] : -1.0e30f;
    }

    // ---- online softmax update (row = q, reduce over 16 lanes of half) ----
#pragma unroll
    for (int r = 0; r < 8; ++r) {
      float mx = fmaxf(sv[0][r], sv[1][r]);
#pragma unroll
      for (int off = 1; off < 16; off <<= 1) mx = fmaxf(mx, __shfl_xor(mx, off, 16));
      float mnew = fmaxf(mrun[r], mx);
      float corr = __expf(mrun[r] - mnew);
      float p0 = __expf(sv[0][r] - mnew);
      float p1 = __expf(sv[1][r] - mnew);
      float rs = p0 + p1;
#pragma unroll
      for (int off = 1; off < 16; off <<= 1) rs += __shfl_xor(rs, off, 16);
      lrun[r] = lrun[r] * corr + rs;
      mrun[r] = mnew;
#pragma unroll
      for (int dt = 0; dt < 4; ++dt) acc[dt][r] *= corr;
      pbuf[wave][r + h8][lm]      = (__bf16)p0;
      pbuf[wave][r + h8][16 + lm] = (__bf16)p1;
    }
    wait_asynccnt0();   // my wave's V-tile async writes landed
    __syncthreads();

    // ---- P (A-fragment) x V (B-fragments from shared LDS V tile) ----
    {
      const __bf16* p = &pbuf[wave][lm][0];
      v16bf pa = ld_frag2(p + h8, p + 16 + h8);
#pragma unroll
      for (int dt = 0; dt < 4; ++dt) {
        const __bf16* vp = &vlds[dt * 16 + lm][0];
        v16bf vf = ld_frag2(vp + h8, vp + 16 + h8);
        acc[dt] = __builtin_amdgcn_wmma_f32_16x16x32_bf16(
            false, pa, false, vf, (short)0, acc[dt], false, false);
      }
    }
    __syncthreads();
  }

  // ---- normalize and write attn output as (b, s, e) bf16 ----
#pragma unroll
  for (int dt = 0; dt < 4; ++dt)
#pragma unroll
    for (int r = 0; r < 8; ++r) {
      int q = qbase + r + h8;
      int d = dt * 16 + lm;
      float val = acc[dt][r] / lrun[r];
      Out[((size_t)(b * SEQ + q) * NHEADS + head) * HD + d] = (__bf16)val;
    }
}

// ---------------------------------------------------------------------------
// Launch
// ---------------------------------------------------------------------------
extern "C" void kernel_launch(void* const* d_in, const int* in_sizes, int n_in,
                              void* d_out, int out_size, void* d_ws, size_t ws_size,
                              hipStream_t stream) {
  const float* x  = (const float*)d_in[0];
  const float* kg = (const float*)d_in[1];
  const float* Wq = (const float*)d_in[2];
  const float* bq = (const float*)d_in[3];
  const float* Wk = (const float*)d_in[4];
  const float* bk = (const float*)d_in[5];
  const float* Wv = (const float*)d_in[6];
  const float* bv = (const float*)d_in[7];
  const float* Wo = (const float*)d_in[8];
  const float* bo = (const float*)d_in[9];
  float* out = (float*)d_out;

  char* ws = (char*)d_ws;
  // workspace carve (49 MiB total)
  __bf16*   xb    = (__bf16*)(ws + 0);                  //  8 MiB  x bf16
  __bf16*   wqb   = (__bf16*)(ws + (8u  << 20));        //  2 MiB
  __bf16*   wkb   = (__bf16*)(ws + (10u << 20));        //  2 MiB
  __bf16*   wvb   = (__bf16*)(ws + (12u << 20));        //  2 MiB
  __bf16*   wob   = (__bf16*)(ws + (14u << 20));        //  2 MiB
  __bf16*   qb    = (__bf16*)(ws + (16u << 20));        //  8 MiB  (bh,s,hd)
  __bf16*   kbuf  = (__bf16*)(ws + (24u << 20));        //  8 MiB  (bh,s,hd)
  __bf16*   vtb   = (__bf16*)(ws + (32u << 20));        //  8 MiB  (bh,hd,s)
  __bf16*   ab    = (__bf16*)(ws + (40u << 20));        //  8 MiB  (b,s,e)
  unsigned* maskw = (unsigned*)(ws + (48u << 20));      //  1 MiB  packed mask

  int n;
  n = MTOT * EMBED;
  cvt_bf16_kernel<<<(n + 255) / 256, 256, 0, stream>>>(x, xb, n);
  n = EMBED * EMBED;
  cvt_bf16_kernel<<<(n + 255) / 256, 256, 0, stream>>>(Wq, wqb, n);
  cvt_bf16_kernel<<<(n + 255) / 256, 256, 0, stream>>>(Wk, wkb, n);
  cvt_bf16_kernel<<<(n + 255) / 256, 256, 0, stream>>>(Wv, wvb, n);
  cvt_bf16_kernel<<<(n + 255) / 256, 256, 0, stream>>>(Wo, wob, n);
  n = BATCH * SEQ * SEQ;
  mask_pack_kernel<<<(n + 255) / 256, 256, 0, stream>>>(kg, maskw, n);

  dim3 gg(MTOT / 128, EMBED / 128);   // 32 x 8
  gemm_bf16_kernel<0><<<gg, 256, 0, stream>>>(xb, wqb, bq, 0.125f, (void*)qb);   // Q pre-scaled
  gemm_bf16_kernel<0><<<gg, 256, 0, stream>>>(xb, wkb, bk, 1.0f,   (void*)kbuf);
  gemm_bf16_kernel<2><<<gg, 256, 0, stream>>>(xb, wvb, bv, 1.0f,   (void*)vtb);

  dim3 ga(SEQ / 64, BATCH * NHEADS);  // 32 x 32
  attn_kernel<<<ga, 128, 0, stream>>>(qb, kbuf, vtb, maskw, ab);

  gemm_bf16_kernel<3><<<gg, 256, 0, stream>>>(ab, wob, bo, 1.0f, (void*)out);
}
